// HeteroGNN_85057532330279
// MI455X (gfx1250) — matrix operand order repaired and checked by reading, not compiled, not run on hardware
//
#include <hip/hip_runtime.h>
#include <hip/hip_bf16.h>
#include <stddef.h>

// ---------------- problem constants (match reference) ----------------
#define NPL 50000
#define NTR 200000
#define NAR 20000
#define DIN 128
#define DH  256
#define DOUT 128
#define EPT 1000000
#define ETP 1000000
#define EAT 400000
#define ETA 400000

typedef __attribute__((ext_vector_type(16))) __bf16 v16bf;
typedef __attribute__((ext_vector_type(8)))  float  v8f;

// ---------------- bf16 hi/lo split helpers ----------------
__device__ __forceinline__ __bf16 us_to_bf(unsigned short s) {
    union { unsigned short u; __bf16 b; } v; v.u = s; return v.b;
}
__device__ __forceinline__ unsigned short f32_to_bf_rne(float f) {
    unsigned int u = __float_as_uint(f);
    unsigned int r = (u + 0x7fffu + ((u >> 16) & 1u)) >> 16;
    return (unsigned short)r;
}
__device__ __forceinline__ void store_split(unsigned short* hi, unsigned short* lo, float v) {
    unsigned short h = f32_to_bf_rne(v);
    float hf = __uint_as_float(((unsigned int)h) << 16);
    *hi = h;
    *lo = f32_to_bf_rne(v - hf);
}
__device__ __forceinline__ void split16(const float* x, v16bf& hi, v16bf& lo) {
#pragma unroll
    for (int j = 0; j < 16; ++j) {
        unsigned short h = f32_to_bf_rne(x[j]);
        float hf = __uint_as_float(((unsigned int)h) << 16);
        unsigned short l = f32_to_bf_rne(x[j] - hf);
        hi[j] = us_to_bf(h);
        lo[j] = us_to_bf(l);
    }
}
__device__ __forceinline__ v16bf load_v16bf(const unsigned short* p) {  // 32B aligned
    union { uint4 q[2]; v16bf v; } u;
    u.q[0] = ((const uint4*)p)[0];
    u.q[1] = ((const uint4*)p)[1];
    return u.v;
}

// A fragment: 16x32 bf16 layout: lane<16 -> K = kb+0..15,
// lane>=16 -> K = kb+16..31; row = lane&15.  On-the-fly f32->bf16 hi/lo
// split with the per-row mean scale folded in.
__device__ __forceinline__ void load_a_frag(const float* __restrict__ A, int ldk,
                                            int row, int kb, int half, float scale,
                                            v16bf& hi, v16bf& lo) {
    const float* p = A + (size_t)row * ldk + kb + half * 16;
    float x[16];
#pragma unroll
    for (int j = 0; j < 8; ++j) x[j]     = p[j]      * scale;
#pragma unroll
    for (int j = 0; j < 8; ++j) x[8 + j] = p[8 + j] * scale;
    split16(x, hi, lo);
}

// ---------------- fused multi-term WMMA GEMM ----------------
// out[M,N] = relu?( sum_t scale_t(row) * A_t[M,K] @ W_t[N,K]^T + bias[N] )
// scale_t(row) = alpha_t / max(cnt_t[row],1)  (cnt nullable -> alpha_t)
// Weights pre-staged as bf16 hi/lo in fragment-friendly (n-major, k-contiguous) order.
// Column tiles are processed in PAIRS with interleaved accumulators so consecutive
// WMMAs never share a D register -> hazard gaps filled with useful work, not v_nop.
__global__ void __launch_bounds__(256)
wmma_gemm3(const float* A0, const unsigned short* H0, const unsigned short* L0,
           const float* C0, float al0,
           const float* A1, const unsigned short* H1, const unsigned short* L1,
           const float* C1, float al1,
           const float* A2, const unsigned short* H2, const unsigned short* L2,
           const float* C2, float al2,
           const float* __restrict__ bias, float* __restrict__ out,
           int M, int N, int K, int relu)
{
    const int lane = threadIdx.x & 31;
    const int wave = threadIdx.x >> 5;
    const int half = lane >> 4;
    const int l15  = lane & 15;
    const int row0 = blockIdx.x * 128 + wave * 16;   // 16-row strip per wave
    if (row0 >= M) return;                            // wave-uniform exit (EXEC stays full)
    const int col0 = blockIdx.y * 128;                // 8 x 16-col tiles per wave

    v8f c[8] = {};

#pragma unroll 1
    for (int t = 0; t < 3; ++t) {
        const float*          A  = (t == 0) ? A0 : (t == 1) ? A1 : A2;
        const unsigned short* WH = (t == 0) ? H0 : (t == 1) ? H1 : H2;
        const unsigned short* WL = (t == 0) ? L0 : (t == 1) ? L1 : L2;
        const float*          CN = (t == 0) ? C0 : (t == 1) ? C1 : C2;
        float alpha              = (t == 0) ? al0 : (t == 1) ? al1 : al2;
        if (A == nullptr) continue;
        float scale = alpha;
        if (CN) scale = alpha / fmaxf(CN[row0 + l15], 1.0f);

        for (int kb = 0; kb < K; kb += 32) {
            v16bf ahi, alo;
            load_a_frag(A, K, row0 + l15, kb, half, scale, ahi, alo);
            // B fragment: 32x16 layout: lane<16 -> K = kb+0..15, lane>=16 -> kb+16..31
            size_t boff = (size_t)(col0 + l15) * K + kb + half * 16;
#pragma unroll
            for (int nt = 0; nt < 8; nt += 2) {
                v16bf bhi0 = load_v16bf(WH + boff + (size_t)(nt + 0) * 16 * K);
                v16bf blo0 = load_v16bf(WL + boff + (size_t)(nt + 0) * 16 * K);
                v16bf bhi1 = load_v16bf(WH + boff + (size_t)(nt + 1) * 16 * K);
                v16bf blo1 = load_v16bf(WL + boff + (size_t)(nt + 1) * 16 * K);
                // fp32-accurate product via bf16 split (hh + hl + lh),
                // alternating accumulators to break D-register dependency chains
                c[nt + 0] = __builtin_amdgcn_wmma_f32_16x16x32_bf16(
                    false, ahi, false, bhi0, (short)0, c[nt + 0], false, false);
                c[nt + 1] = __builtin_amdgcn_wmma_f32_16x16x32_bf16(
                    false, ahi, false, bhi1, (short)0, c[nt + 1], false, false);
                c[nt + 0] = __builtin_amdgcn_wmma_f32_16x16x32_bf16(
                    false, ahi, false, blo0, (short)0, c[nt + 0], false, false);
                c[nt + 1] = __builtin_amdgcn_wmma_f32_16x16x32_bf16(
                    false, ahi, false, blo1, (short)0, c[nt + 1], false, false);
                c[nt + 0] = __builtin_amdgcn_wmma_f32_16x16x32_bf16(
                    false, alo, false, bhi0, (short)0, c[nt + 0], false, false);
                c[nt + 1] = __builtin_amdgcn_wmma_f32_16x16x32_bf16(
                    false, alo, false, bhi1, (short)0, c[nt + 1], false, false);
            }
        }
    }

    // C/D layout: VGPR i -> row = row0 + half*8 + i, col = col0 + nt*16 + (lane&15)
#pragma unroll
    for (int nt = 0; nt < 8; ++nt) {
        int col = col0 + nt * 16 + l15;
        float b = bias ? bias[col] : 0.0f;
#pragma unroll
        for (int i = 0; i < 8; ++i) {
            int row = row0 + half * 8 + i;
            float v = c[nt][i] + b;
            if (relu) v = fmaxf(v, 0.0f);
            out[(size_t)row * N + col] = v;
        }
    }
}

// ---------------- edge scatter (segment sum + counts) ----------------
// one wave per edge; D in {128,256}; float4 gathers + f32 global atomics
__global__ void __launch_bounds__(256)
scatter_sum(const float* __restrict__ x, const int* __restrict__ esrc,
            const int* __restrict__ edst, int E, int D,
            float* __restrict__ sum, float* __restrict__ cnt)
{
    int gid  = blockIdx.x * blockDim.x + threadIdx.x;
    int lane = gid & 31;
    int wid  = gid >> 5;
    int nw   = (gridDim.x * blockDim.x) >> 5;
    int chunks = D >> 7;  // 128 floats per wave-pass
    for (int e = wid; e < E; e += nw) {
        int s = esrc[e], d = edst[e];
        const float4* xs = (const float4*)(x + (size_t)s * D);
        float* sd = sum + (size_t)d * D;
        for (int cch = 0; cch < chunks; ++cch) {
            float4 v = xs[cch * 32 + lane];
            int base = (cch * 32 + lane) * 4;
            atomicAdd(sd + base + 0, v.x);
            atomicAdd(sd + base + 1, v.y);
            atomicAdd(sd + base + 2, v.z);
            atomicAdd(sd + base + 3, v.w);
        }
        if (lane == 0 && cnt) atomicAdd(cnt + d, 1.0f);
    }
}

// ---------------- small utility kernels ----------------
__global__ void __launch_bounds__(256)
zero_f4(float4* p, size_t n4)
{
    size_t i = (size_t)blockIdx.x * blockDim.x + threadIdx.x;
    size_t stride = (size_t)gridDim.x * blockDim.x;
    float4 z = {0.f, 0.f, 0.f, 0.f};
    for (; i < n4; i += stride) p[i] = z;
}

__global__ void __launch_bounds__(256)
add2_f4(const float4* a, const float4* b, float4* o, size_t n4)
{
    size_t i = (size_t)blockIdx.x * blockDim.x + threadIdx.x;
    size_t stride = (size_t)gridDim.x * blockDim.x;
    for (; i < n4; i += stride) {
        float4 x = a[i], y = b[i];
        float4 r = {x.x + y.x, x.y + y.y, x.z + y.z, x.w + y.w};
        o[i] = r;
    }
}

// stage a plain fp32 weight matrix (n-major, k-contiguous) into bf16 hi/lo
__global__ void __launch_bounds__(256)
stage_bf16(const float* __restrict__ src, unsigned short* __restrict__ hi,
           unsigned short* __restrict__ lo, int n)
{
    int i = blockIdx.x * blockDim.x + threadIdx.x;
    if (i < n) store_split(hi + i, lo + i, src[i]);
}

// fold hetero-mean 0.5 factors + residual projection into combined weights
// (emitted directly as staged bf16 hi/lo) and combined fp32 biases
__global__ void __launch_bounds__(256)
prep_weights(const float* W1r, const float* b1,
             const float* W2r, const float* b2,
             const float* Wres, const float* bres,
             unsigned short* wt1r_h, unsigned short* wt1r_l,
             unsigned short* wt2r_h, unsigned short* wt2r_l,
             unsigned short* wp2r_h, unsigned short* wp2r_l,
             unsigned short* wa2r_h, unsigned short* wa2r_l,
             float* bt1, float* bt2, float* bp2, float* ba2)
{
    int i = blockIdx.x * blockDim.x + threadIdx.x;
    const int S1 = DH * DIN;    // 32768
    const int S2 = DOUT * DH;   // 32768
    if (i < S1) store_split(wt1r_h + i, wt1r_l + i,
                            0.5f * (W1r[0 * S1 + i] + W1r[2 * S1 + i]));
    if (i < S2) {
        store_split(wt2r_h + i, wt2r_l + i,
                    0.5f * (W2r[0 * S2 + i] + W2r[2 * S2 + i]) + Wres[1 * S2 + i]);
        store_split(wp2r_h + i, wp2r_l + i, W2r[1 * S2 + i] + Wres[0 * S2 + i]);
        store_split(wa2r_h + i, wa2r_l + i, W2r[3 * S2 + i] + Wres[2 * S2 + i]);
    }
    if (i < DH)   bt1[i] = 0.5f * (b1[0 * DH + i] + b1[2 * DH + i]);
    if (i < DOUT) {
        bt2[i] = 0.5f * (b2[0 * DOUT + i] + b2[2 * DOUT + i]) + bres[1 * DOUT + i];
        bp2[i] = b2[1 * DOUT + i] + bres[0 * DOUT + i];
        ba2[i] = b2[3 * DOUT + i] + bres[2 * DOUT + i];
    }
}

// ---------------- host-side launch ----------------
static inline unsigned cdiv(unsigned a, unsigned b) { return (a + b - 1) / b; }

extern "C" void kernel_launch(void* const* d_in, const int* in_sizes, int n_in,
                              void* d_out, int out_size, void* d_ws, size_t ws_size,
                              hipStream_t stream)
{
    (void)in_sizes; (void)n_in; (void)out_size; (void)ws_size;
    const float* x_p   = (const float*)d_in[0];
    const float* x_t   = (const float*)d_in[1];
    const float* x_a   = (const float*)d_in[2];
    const float* emb_p = (const float*)d_in[3];
    const float* emb_a = (const float*)d_in[4];
    const float* W1l   = (const float*)d_in[5];
    const float* b1    = (const float*)d_in[6];
    const float* W1r   = (const float*)d_in[7];
    const float* W2l   = (const float*)d_in[8];
    const float* b2    = (const float*)d_in[9];
    const float* W2r   = (const float*)d_in[10];
    const float* Wres  = (const float*)d_in[11];
    const float* bres  = (const float*)d_in[12];
    const int* e_pt = (const int*)d_in[13];
    const int* e_tp = (const int*)d_in[14];
    const int* e_at = (const int*)d_in[15];
    const int* e_ta = (const int*)d_in[16];
    float* out = (float*)d_out;
    float* ws  = (float*)d_ws;

    // ---- workspace layout (floats) ----
    // Region A (size RA): conv1 {xp, xa, sum_pt, sum_at, sum_tp, sum_ta},
    //                     later aliased by conv2 {s2_pt, s2_at, s2_tp, s2_ta}
    const size_t RA = (size_t)NTR * DH * 2 + (size_t)NPL * DH + (size_t)NAR * DH; // 120.32M
    const size_t o_xp = 0;
    const size_t o_xa = o_xp + (size_t)NPL * DIN;
    const size_t o_s_pt = o_xa + (size_t)NAR * DIN;
    const size_t o_s_at = o_s_pt + (size_t)NTR * DIN;
    const size_t o_s_tp = o_s_at + (size_t)NTR * DIN;
    const size_t o_s_ta = o_s_tp + (size_t)NPL * DIN;
    const size_t conv1_sum_floats = (size_t)NTR * DIN * 2 + (size_t)NPL * DIN + (size_t)NAR * DIN;
    // conv2 aliases region A:
    const size_t o2_pt = 0;
    const size_t o2_at = o2_pt + (size_t)NTR * DH;
    const size_t o2_tp = o2_at + (size_t)NTR * DH;
    const size_t o2_ta = o2_tp + (size_t)NPL * DH;
    // Region B: counts (persist across both layers)
    const size_t o_c_pt = RA;
    const size_t o_c_at = o_c_pt + NTR;
    const size_t o_c_tp = o_c_at + NTR;
    const size_t o_c_ta = o_c_tp + NPL;
    const size_t o_c_end = o_c_ta + NAR;           // +470000
    // Region C: layer-1 activations
    const size_t o_p1 = o_c_end;
    const size_t o_t1 = o_p1 + (size_t)NPL * DH;
    const size_t o_a1 = o_t1 + (size_t)NTR * DH;
    const size_t o_h_end = o_a1 + (size_t)NAR * DH;
    // Region D: combined fp32 biases
    const size_t o_bt1 = o_h_end;
    const size_t o_bt2 = o_bt1 + DH;
    const size_t o_bp2 = o_bt2 + DOUT;
    const size_t o_ba2 = o_bp2 + DOUT;
    // Region E: staged bf16 hi/lo weights (ushort units), 32B-aligned
    const size_t o_stg_f = (o_ba2 + DOUT + 15) & ~(size_t)15;
    unsigned short* stg = (unsigned short*)(ws + o_stg_f);
    const size_t S = (size_t)DH * DIN;             // 32768 (== DOUT*DH)
    unsigned short* w1l_h  = stg;                  // 4 slices, hi then lo
    unsigned short* w1l_l  = stg + 4 * S;
    unsigned short* w2l_h  = stg + 8 * S;
    unsigned short* w2l_l  = stg + 12 * S;
    unsigned short* w1r1_h = stg + 16 * S;  unsigned short* w1r1_l = stg + 17 * S;
    unsigned short* w1r3_h = stg + 18 * S;  unsigned short* w1r3_l = stg + 19 * S;
    unsigned short* wt1r_h = stg + 20 * S;  unsigned short* wt1r_l = stg + 21 * S;
    unsigned short* wt2r_h = stg + 22 * S;  unsigned short* wt2r_l = stg + 23 * S;
    unsigned short* wp2r_h = stg + 24 * S;  unsigned short* wp2r_l = stg + 25 * S;
    unsigned short* wa2r_h = stg + 26 * S;  unsigned short* wa2r_l = stg + 27 * S;

    // ---- 1) zero conv1 sums + counts ----
    zero_f4<<<2048, 256, 0, stream>>>((float4*)(ws + o_s_pt), conv1_sum_floats / 4);
    zero_f4<<<64,   256, 0, stream>>>((float4*)(ws + o_c_pt), (size_t)(2 * NTR + NPL + NAR) / 4);

    // ---- 2) id-embedding adds: xp = x_p + emb_p ; xa = x_a + emb_a ----
    add2_f4<<<2048, 256, 0, stream>>>((const float4*)x_p, (const float4*)emb_p,
                                      (float4*)(ws + o_xp), (size_t)NPL * DIN / 4);
    add2_f4<<<1024, 256, 0, stream>>>((const float4*)x_a, (const float4*)emb_a,
                                      (float4*)(ws + o_xa), (size_t)NAR * DIN / 4);

    // ---- 3) stage weights as bf16 hi/lo (one-time, ~1.75 MB, L2-resident) ----
    stage_bf16<<<cdiv(4 * S, 256), 256, 0, stream>>>(W1l, w1l_h, w1l_l, 4 * S);
    stage_bf16<<<cdiv(4 * S, 256), 256, 0, stream>>>(W2l, w2l_h, w2l_l, 4 * S);
    stage_bf16<<<cdiv(S, 256), 256, 0, stream>>>(W1r + 1 * S, w1r1_h, w1r1_l, S);
    stage_bf16<<<cdiv(S, 256), 256, 0, stream>>>(W1r + 3 * S, w1r3_h, w1r3_l, S);
    prep_weights<<<cdiv(S, 256), 256, 0, stream>>>(
        W1r, b1, W2r, b2, Wres, bres,
        wt1r_h, wt1r_l, wt2r_h, wt2r_l, wp2r_h, wp2r_l, wa2r_h, wa2r_l,
        ws + o_bt1, ws + o_bt2, ws + o_bp2, ws + o_ba2);

    // ---- 4) conv1 edge aggregation (segment sums + counts) ----
    scatter_sum<<<4096, 256, 0, stream>>>(ws + o_xp, e_pt, e_pt + EPT, EPT, DIN,
                                          ws + o_s_pt, ws + o_c_pt);
    scatter_sum<<<2048, 256, 0, stream>>>(ws + o_xa, e_at, e_at + EAT, EAT, DIN,
                                          ws + o_s_at, ws + o_c_at);
    scatter_sum<<<4096, 256, 0, stream>>>(x_t, e_tp, e_tp + ETP, ETP, DIN,
                                          ws + o_s_tp, ws + o_c_tp);
    scatter_sum<<<2048, 256, 0, stream>>>(x_t, e_ta, e_ta + ETA, ETA, DIN,
                                          ws + o_s_ta, ws + o_c_ta);

    // ---- 5) conv1 fused WMMA GEMMs (mean-scale + bias + relu fused) ----
    // t1 = relu(0.5*agg_pt@W1l0^T + 0.5*agg_at@W1l2^T + xt@(0.5(W1r0+W1r2))^T + 0.5(b10+b12))
    wmma_gemm3<<<dim3(cdiv(NTR, 128), DH / 128), 256, 0, stream>>>(
        ws + o_s_pt, w1l_h + 0 * S, w1l_l + 0 * S, ws + o_c_pt, 0.5f,
        ws + o_s_at, w1l_h + 2 * S, w1l_l + 2 * S, ws + o_c_at, 0.5f,
        x_t,         wt1r_h,        wt1r_l,        nullptr,     1.0f,
        ws + o_bt1, ws + o_t1, NTR, DH, DIN, 1);
    // p1 = relu(agg_tp@W1l1^T + xp@W1r1^T + b1[1])
    wmma_gemm3<<<dim3(cdiv(NPL, 128), DH / 128), 256, 0, stream>>>(
        ws + o_s_tp, w1l_h + 1 * S, w1l_l + 1 * S, ws + o_c_tp, 1.0f,
        ws + o_xp,   w1r1_h,        w1r1_l,        nullptr,     1.0f,
        nullptr, nullptr, nullptr, nullptr, 0.0f,
        b1 + 1 * DH, ws + o_p1, NPL, DH, DIN, 1);
    // a1 = relu(agg_ta@W1l3^T + xa@W1r3^T + b1[3])
    wmma_gemm3<<<dim3(cdiv(NAR, 128), DH / 128), 256, 0, stream>>>(
        ws + o_s_ta, w1l_h + 3 * S, w1l_l + 3 * S, ws + o_c_ta, 1.0f,
        ws + o_xa,   w1r3_h,        w1r3_l,        nullptr,     1.0f,
        nullptr, nullptr, nullptr, nullptr, 0.0f,
        b1 + 3 * DH, ws + o_a1, NAR, DH, DIN, 1);

    // ---- 6) zero conv2 sums (reuse region A; counts are reused, not rezeroed) ----
    zero_f4<<<4096, 256, 0, stream>>>((float4*)(ws + o2_pt), RA / 4);

    // ---- 7) conv2 edge aggregation over h1 (D = 256; same counts as conv1) ----
    scatter_sum<<<4096, 256, 0, stream>>>(ws + o_p1, e_pt, e_pt + EPT, EPT, DH,
                                          ws + o2_pt, nullptr);
    scatter_sum<<<2048, 256, 0, stream>>>(ws + o_a1, e_at, e_at + EAT, EAT, DH,
                                          ws + o2_at, nullptr);
    scatter_sum<<<4096, 256, 0, stream>>>(ws + o_t1, e_tp, e_tp + ETP, ETP, DH,
                                          ws + o2_tp, nullptr);
    scatter_sum<<<2048, 256, 0, stream>>>(ws + o_t1, e_ta, e_ta + ETA, ETA, DH,
                                          ws + o2_ta, nullptr);

    // ---- 8) conv2 + residual fused WMMA GEMMs -> d_out (p, t, a concatenated) ----
    float* out_p = out;
    float* out_t = out + (size_t)NPL * DOUT;
    float* out_a = out + (size_t)(NPL + NTR) * DOUT;
    // t_out = 0.5*agg2_pt@W2l0^T + 0.5*agg2_at@W2l2^T + t1@(0.5W2r0+0.5W2r2+Wres1)^T + bt2
    wmma_gemm3<<<dim3(cdiv(NTR, 128), DOUT / 128), 256, 0, stream>>>(
        ws + o2_pt, w2l_h + 0 * S, w2l_l + 0 * S, ws + o_c_pt, 0.5f,
        ws + o2_at, w2l_h + 2 * S, w2l_l + 2 * S, ws + o_c_at, 0.5f,
        ws + o_t1,  wt2r_h,        wt2r_l,        nullptr,     1.0f,
        ws + o_bt2, out_t, NTR, DOUT, DH, 0);
    // p_out = agg2_tp@W2l1^T + p1@(W2r1+Wres0)^T + bp2
    wmma_gemm3<<<dim3(cdiv(NPL, 128), DOUT / 128), 256, 0, stream>>>(
        ws + o2_tp, w2l_h + 1 * S, w2l_l + 1 * S, ws + o_c_tp, 1.0f,
        ws + o_p1,  wp2r_h,        wp2r_l,        nullptr,     1.0f,
        nullptr, nullptr, nullptr, nullptr, 0.0f,
        ws + o_bp2, out_p, NPL, DOUT, DH, 0);
    // a_out = agg2_ta@W2l3^T + a1@(W2r3+Wres2)^T + ba2
    wmma_gemm3<<<dim3(cdiv(NAR, 128), DOUT / 128), 256, 0, stream>>>(
        ws + o2_ta, w2l_h + 3 * S, w2l_l + 3 * S, ws + o_c_ta, 1.0f,
        ws + o_a1,  wa2r_h,        wa2r_l,        nullptr,     1.0f,
        nullptr, nullptr, nullptr, nullptr, 0.0f,
        ws + o_ba2, out_a, NAR, DOUT, DH, 0);
}
